// SelfSpatialAttention3D_42167988912599
// MI455X (gfx1250) — matrix operand order repaired and checked by reading, not compiled, not run on hardware
//
#include <hip/hip_runtime.h>

typedef __attribute__((ext_vector_type(16))) _Float16 v16h;
typedef __attribute__((ext_vector_type(8)))  _Float16 v8h;
typedef __attribute__((ext_vector_type(8)))  float    v8f;

union V16U { v16h v; v8h h[2]; };

#define EPS 1e-3f

#if defined(__gfx1250__) && __has_builtin(__builtin_amdgcn_global_load_async_to_lds_b128) && __has_builtin(__builtin_amdgcn_s_wait_asynccnt)
#define HAVE_ASYNC 1
#else
#define HAVE_ASYNC 0
#endif

#if HAVE_ASYNC
// the builtin's pointer params are GCC-style vector_size(16) int4 in AS1/AS3
typedef int v4i_vs __attribute__((vector_size(16)));
#define AS1(p) ((__attribute__((address_space(1))) v4i_vs*)(p))
#define AS3(p) ((__attribute__((address_space(3))) v4i_vs*)(p))
#endif

// ---- problem sizes ----
// B=8, N=16384 (=Z*H*W), C=512, C8=64
static constexpr int  CBIG = 512;
static constexpr int  C8   = 64;
static constexpr int  NTOT = 16384;
static constexpr int  BAT  = 8;

// ---- workspace layout (bytes) ----
static constexpr size_t OFF_WQKV = 0;                          // 16*12*32*16 f16 = 196608
static constexpr size_t OFF_WP   = 196608;                     // 2*32*32*16 f16 = 65536
static constexpr size_t OFF_BQKV = 262144;                     // 192 f32
static constexpr size_t OFF_BP2  = 263168;                     // 512 f32
static constexpr size_t OFF_S    = 265216;                     // 8*64*64 f32 = 131072
static constexpr size_t OFF_BETA = 396288;                     // 8*64*64 f16 = 65536
static constexpr size_t OFF_QT   = 462848;                     // 8*64*16384 f16 = 16 MB
static constexpr size_t OFF_KT   = OFF_QT  + 16777216;
static constexpr size_t OFF_V    = OFF_KT  + 16777216;
static constexpr size_t OFF_ATTN = OFF_V   + 16777216;

__device__ __forceinline__ v8f wmma16x16x32(v16h a, v16h b, v8f c) {
  return __builtin_amdgcn_wmma_f32_16x16x32_f16(false, a, false, b, (short)0, c, false, false);
}

// =====================================================================
// Kernel 0: fold BN1 into wq/wk/wv (+bias), BN2 into wp (+bias),
//           pre-swizzle weights into WMMA B-fragment layout, zero s.
// =====================================================================
__global__ void k_prep(const float* g1, const float* b1, const float* m1, const float* v1,
                       const float* wq, const float* bq, const float* wk, const float* bk,
                       const float* wv, const float* bv, const float* wp, const float* bp,
                       const float* g2, const float* b2, const float* m2, const float* v2,
                       _Float16* wqkv_frag, _Float16* wp_frag, float* bqkv, float* bp2,
                       float* sbuf)
{
  const int tid = blockIdx.x * blockDim.x + threadIdx.x;
  const int nth = gridDim.x * blockDim.x;

  // QKV B-fragments: [kt(16)][w(12)][lane(32)][j(16)]
  for (int idx = tid; idx < 16 * 12 * 32 * 16; idx += nth) {
    int j = idx & 15; int t = idx >> 4;
    int lane = t & 31; t >>= 5;
    int w = t % 12; int kt = t / 12;
    int c = kt * 32 + ((lane < 16) ? j : 16 + j);      // K index (input channel)
    int d = (w & 3) * 16 + (lane & 15);                // N index (output channel)
    const float* W = (w < 4) ? wq : (w < 8) ? wk : wv;
    float a1 = g1[c] * rsqrtf(v1[c] + EPS);
    wqkv_frag[idx] = (_Float16)(W[c * C8 + d] * a1);
  }
  // wp B-fragments: [kt(2)][ct(32)][lane(32)][j(16)]
  for (int idx = tid; idx < 2 * 32 * 32 * 16; idx += nth) {
    int j = idx & 15; int t = idx >> 4;
    int lane = t & 31; t >>= 5;
    int ct = t & 31; int kt = t >> 5;
    int k = kt * 32 + ((lane < 16) ? j : 16 + j);
    int c = ct * 16 + (lane & 15);
    float a2 = g2[c] * rsqrtf(v2[c] + EPS);
    wp_frag[idx] = (_Float16)(wp[k * CBIG + c] * a2);
  }
  // folded qkv biases: b'[d] = b[d] + sum_c (beta1-mean1*a1)[c] * W[c,d]
  for (int idx = tid; idx < 192; idx += nth) {
    int which = idx >> 6; int d = idx & 63;
    const float* W  = (which == 0) ? wq : (which == 1) ? wk : wv;
    const float* bb = (which == 0) ? bq : (which == 1) ? bk : bv;
    float s = bb[d];
    for (int c = 0; c < CBIG; ++c) {
      float a1 = g1[c] * rsqrtf(v1[c] + EPS);
      float sh = b1[c] - m1[c] * a1;
      s += sh * W[c * C8 + d];
    }
    bqkv[idx] = s;
  }
  // folded output bias: bp'[c] = bp[c]*a2 + (beta2 - mean2*a2)
  for (int idx = tid; idx < CBIG; idx += nth) {
    float a2 = g2[idx] * rsqrtf(v2[idx] + EPS);
    bp2[idx] = bp[idx] * a2 + (b2[idx] - m2[idx] * a2);
  }
  // zero split-K accumulator
  for (int idx = tid; idx < BAT * C8 * C8; idx += nth) sbuf[idx] = 0.0f;
}

// =====================================================================
// Kernel 1: fused BN1 + q/k/v projections.  64 rows per WG (4 waves).
// Writes qT,kT (transposed, channel-major) and v (natural).
// =====================================================================
#define LDS1 264   // 256 + 8 halfs pad -> 4r bank pattern, conflict free
__global__ __launch_bounds__(128) void k_qkv(const float* __restrict__ x,
                                             const _Float16* __restrict__ wqkv_frag,
                                             const float* __restrict__ bqkv,
                                             _Float16* __restrict__ qT,
                                             _Float16* __restrict__ kT,
                                             _Float16* __restrict__ vN)
{
  __shared__ _Float16 xs[64 * LDS1];
  const int tid  = threadIdx.x;
  const int lane = tid & 31, wid = tid >> 5;
  const long row0 = (long)blockIdx.x * 64;     // global row in (B*N)
  const int  b  = (int)(row0 >> 14);
  const int  n0 = (int)(row0 & (NTOT - 1));
  const int  mrow = lane & 15;
  const int  loA  = (lane < 16) ? 0 : 16;

  const v8f vzero = {0, 0, 0, 0, 0, 0, 0, 0};
  v8f acc[12];
#pragma unroll
  for (int i = 0; i < 12; ++i) acc[i] = vzero;

  for (int half = 0; half < 2; ++half) {
    __syncthreads();
    for (int i = tid; i < 64 * 256; i += 128) {
      int r = i >> 8, c = i & 255;
      xs[r * LDS1 + c] = (_Float16)x[(row0 + r) * CBIG + half * 256 + c];
    }
    if (half == 0) {
      // prefetch the second 64KB half of the x tile into the cache hierarchy
      for (int i = tid; i < 512; i += 128) {
        const int r = i >> 3, c0 = (i & 7) * 32;
        __builtin_prefetch(&x[(row0 + r) * CBIG + 256 + c0], 0, 0);
      }
    }
    __syncthreads();
#pragma unroll
    for (int kt8 = 0; kt8 < 8; ++kt8) {
      const int kt = half * 8 + kt8;
      const _Float16* ap = &xs[(wid * 16 + mrow) * LDS1 + kt8 * 32 + loA];
      V16U ua; ua.h[0] = *(const v8h*)ap; ua.h[1] = *(const v8h*)(ap + 8);
#pragma unroll
      for (int w = 0; w < 12; ++w) {
        const _Float16* bpn = wqkv_frag + (((kt * 12 + w) * 32 + lane) << 4);
        V16U ub; ub.h[0] = *(const v8h*)bpn; ub.h[1] = *(const v8h*)(bpn + 8);
        acc[w] = wmma16x16x32(ua.v, ub.v, acc[w]);
      }
    }
  }

#pragma unroll
  for (int w = 0; w < 12; ++w) {
    const int which = w >> 2, ctl = w & 3;
    const int col  = ctl * 16 + mrow;           // output channel
    const float bias = bqkv[which * 64 + col];
    if (which < 2) {
      _Float16* dst = (which == 0) ? qT : kT;   // [b][i][n] channel-major
      const long nb = (long)b * C8 * NTOT + (long)col * NTOT + n0 + wid * 16 + ((lane < 16) ? 0 : 8);
      v8h o;
#pragma unroll
      for (int r = 0; r < 8; ++r) o[r] = (_Float16)(acc[w][r] + bias);
      *(v8h*)(dst + nb) = o;                    // 16B contiguous transposed store
    } else {
#pragma unroll
      for (int r = 0; r < 8; ++r) {
        const int n = n0 + wid * 16 + r + ((lane < 16) ? 0 : 8);
        vN[(long)b * NTOT * C8 + (long)n * C8 + col] = (_Float16)(acc[w][r] + bias);
      }
    }
  }
}

// =====================================================================
// Kernel 2: s[b,i,j] = sum_n q[n,i]*k[n,j]  (split-K, 32 splits/batch)
// =====================================================================
#define LDS2 136   // 128 + 8 pad
__global__ __launch_bounds__(128) void k_gram(const _Float16* __restrict__ qT,
                                              const _Float16* __restrict__ kT,
                                              float* __restrict__ sbuf)
{
  __shared__ _Float16 qs[64 * LDS2];
  __shared__ _Float16 ks[64 * LDS2];
  const int tid = threadIdx.x, lane = tid & 31, wid = tid >> 5;
  const int b     = blockIdx.x >> 5;
  const int split = blockIdx.x & 31;
  const long base = (long)b * C8 * NTOT;
  const int mrow = lane & 15;
  const int loA  = (lane < 16) ? 0 : 16;
  const int loB  = (lane < 16) ? 0 : 16;

  const v8f vzero = {0, 0, 0, 0, 0, 0, 0, 0};
  v8f acc[4];
#pragma unroll
  for (int i = 0; i < 4; ++i) acc[i] = vzero;

  for (int chunk = 0; chunk < 4; ++chunk) {
    const int n0 = split * 512 + chunk * 128;
    __syncthreads();
#if HAVE_ASYNC
    // pure f16 tile copy -> asynchronous global->LDS DMA path (ASYNCcnt)
    for (int i = tid; i < 64 * 128 / 8; i += 128) {
      int r = i >> 4, c8 = (i & 15) * 8;
      __builtin_amdgcn_global_load_async_to_lds_b128(
          AS1(qT + base + (long)r * NTOT + n0 + c8), AS3(&qs[r * LDS2 + c8]), 0, 0);
      __builtin_amdgcn_global_load_async_to_lds_b128(
          AS1(kT + base + (long)r * NTOT + n0 + c8), AS3(&ks[r * LDS2 + c8]), 0, 0);
    }
    __builtin_amdgcn_s_wait_asynccnt(0);
#else
    for (int i = tid; i < 64 * 128; i += 128) {
      int r = i >> 7, c = i & 127;
      qs[r * LDS2 + c] = qT[base + (long)r * NTOT + n0 + c];
      ks[r * LDS2 + c] = kT[base + (long)r * NTOT + n0 + c];
    }
#endif
    __syncthreads();
#pragma unroll
    for (int kt = 0; kt < 4; ++kt) {
      const _Float16* ap = &qs[(wid * 16 + mrow) * LDS2 + kt * 32 + loA];
      V16U ua; ua.h[0] = *(const v8h*)ap; ua.h[1] = *(const v8h*)(ap + 8);
#pragma unroll
      for (int ct = 0; ct < 4; ++ct) {
        const _Float16* bpt = &ks[(ct * 16 + mrow) * LDS2 + kt * 32 + loB];
        V16U ub; ub.h[0] = *(const v8h*)bpt; ub.h[1] = *(const v8h*)(bpt + 8);
        acc[ct] = wmma16x16x32(ua.v, ub.v, acc[ct]);
      }
    }
  }
#pragma unroll
  for (int ct = 0; ct < 4; ++ct) {
    const int j = ct * 16 + mrow;
#pragma unroll
    for (int r = 0; r < 8; ++r) {
      const int i = wid * 16 + r + ((lane < 16) ? 0 : 8);
      atomicAdd(&sbuf[b * 4096 + i * 64 + j], acc[ct][r]);
    }
  }
}

// =====================================================================
// Kernel 3: row softmax of s; store beta-transposed f16 (A-matrix of attn GEMM)
// =====================================================================
__global__ __launch_bounds__(128) void k_softmax(const float* __restrict__ sbuf,
                                                 _Float16* __restrict__ betaT)
{
  const int tid = threadIdx.x, lane = tid & 31, wid = tid >> 5;
  const int rg = blockIdx.x * 4 + wid;
  const int b = rg >> 6, m = rg & 63;
  const float x0 = sbuf[b * 4096 + m * 64 + lane];
  const float x1 = sbuf[b * 4096 + m * 64 + 32 + lane];
  float mx = fmaxf(x0, x1);
#pragma unroll
  for (int off = 16; off; off >>= 1) mx = fmaxf(mx, __shfl_xor(mx, off, 32));
  const float e0 = __expf(x0 - mx), e1 = __expf(x1 - mx);
  float sum = e0 + e1;
#pragma unroll
  for (int off = 16; off; off >>= 1) sum += __shfl_xor(sum, off, 32);
  const float inv = 1.0f / sum;
  betaT[b * 4096 + lane * 64 + m]        = (_Float16)(e0 * inv);
  betaT[b * 4096 + (lane + 32) * 64 + m] = (_Float16)(e1 * inv);
}

// =====================================================================
// Kernel 4: attn[b,i,n] = sum_m beta[m,i] * v[n,m]   (stored i-major so the
// Keras reshape -> final-GEMM A rows are contiguous)
// =====================================================================
#define LDS4 72   // 64 + 8 pad
__global__ __launch_bounds__(128) void k_attn(const _Float16* __restrict__ vN,
                                              const _Float16* __restrict__ betaT,
                                              _Float16* __restrict__ attn)
{
  __shared__ _Float16 vs[256 * LDS4];
  const int tid = threadIdx.x, lane = tid & 31, wid = tid >> 5;
  const int b  = blockIdx.x >> 6;
  const int n0 = (blockIdx.x & 63) * 256;
#if HAVE_ASYNC
  const long vbase = (long)b * NTOT * C8 + (long)n0 * C8;
  for (int i = tid; i < 256 * 64 / 8; i += 128) {
    int nl = i >> 3, m8 = (i & 7) * 8;
    __builtin_amdgcn_global_load_async_to_lds_b128(
        AS1(vN + vbase + (long)nl * C8 + m8), AS3(&vs[nl * LDS4 + m8]), 0, 0);
  }
  __builtin_amdgcn_s_wait_asynccnt(0);
#else
  for (int i = tid; i < 256 * 64; i += 128) {
    int nl = i >> 6, m = i & 63;
    vs[nl * LDS4 + m] = vN[(long)b * NTOT * C8 + (long)(n0 + nl) * C8 + m];
  }
#endif
  __syncthreads();
  const int mrow = lane & 15;
  const int loA  = (lane < 16) ? 0 : 16;
  const int loB  = (lane < 16) ? 0 : 16;

  const v8f vzero = {0, 0, 0, 0, 0, 0, 0, 0};
  v8f acc[4][4];
#pragma unroll
  for (int i = 0; i < 4; ++i)
#pragma unroll
    for (int j = 0; j < 4; ++j) acc[i][j] = vzero;

#pragma unroll
  for (int kt = 0; kt < 2; ++kt) {
    v16h A[4];
#pragma unroll
    for (int rt = 0; rt < 4; ++rt) {
      const _Float16* ap = betaT + b * 4096 + (rt * 16 + mrow) * 64 + kt * 32 + loA;
      V16U u; u.h[0] = *(const v8h*)ap; u.h[1] = *(const v8h*)(ap + 8);
      A[rt] = u.v;
    }
#pragma unroll
    for (int ctl = 0; ctl < 4; ++ctl) {
      const int nloc = wid * 64 + ctl * 16 + mrow;
      const _Float16* bpt = &vs[nloc * LDS4 + kt * 32 + loB];
      V16U ub; ub.h[0] = *(const v8h*)bpt; ub.h[1] = *(const v8h*)(bpt + 8);
#pragma unroll
      for (int rt = 0; rt < 4; ++rt) acc[rt][ctl] = wmma16x16x32(A[rt], ub.v, acc[rt][ctl]);
    }
  }
#pragma unroll
  for (int rt = 0; rt < 4; ++rt)
#pragma unroll
    for (int ctl = 0; ctl < 4; ++ctl) {
      const int ng = n0 + wid * 64 + ctl * 16 + mrow;
#pragma unroll
      for (int r = 0; r < 8; ++r) {
        const int i = rt * 16 + r + ((lane < 16) ? 0 : 8);
        attn[(long)b * (C8 * NTOT) + (long)i * NTOT + ng] = (_Float16)acc[rt][ctl][r];
      }
    }
}

// =====================================================================
// Kernel 5: output projection + folded BN2.  64 rows/WG, 8 waves.
// =====================================================================
__global__ __launch_bounds__(256) void k_proj(const _Float16* __restrict__ attn,
                                              const _Float16* __restrict__ wp_frag,
                                              const float* __restrict__ bp2,
                                              float* __restrict__ out)
{
  const int tid = threadIdx.x, lane = tid & 31, wid = tid >> 5;
  const long row0 = (long)blockIdx.x * 64;
  const int  b   = (int)(row0 >> 14);
  const long mm0 = row0 & (NTOT - 1);
  const int  rt  = wid >> 1;
  const int  cth = (wid & 1) * 16;
  const int  mrow = lane & 15;
  const int  loA  = (lane < 16) ? 0 : 16;

  v16h A[2];
#pragma unroll
  for (int kt = 0; kt < 2; ++kt) {
    const _Float16* ap = attn + (long)b * (C8 * NTOT) + (mm0 + rt * 16 + mrow) * 64 + kt * 32 + loA;
    V16U u; u.h[0] = *(const v8h*)ap; u.h[1] = *(const v8h*)(ap + 8);
    A[kt] = u.v;
  }
  const v8f vzero = {0, 0, 0, 0, 0, 0, 0, 0};
#pragma unroll
  for (int ci = 0; ci < 16; ++ci) {
    const int ct = cth + ci;
    v8f acc = vzero;
#pragma unroll
    for (int kt = 0; kt < 2; ++kt) {
      const _Float16* bpt = wp_frag + (((kt * 32 + ct) * 32 + lane) << 4);
      V16U ub; ub.h[0] = *(const v8h*)bpt; ub.h[1] = *(const v8h*)(bpt + 8);
      acc = wmma16x16x32(A[kt], ub.v, acc);
    }
    const int c = ct * 16 + mrow;
    const float bias = bp2[c];
#pragma unroll
    for (int r = 0; r < 8; ++r) {
      const long mg = row0 + rt * 16 + r + ((lane < 16) ? 0 : 8);
      out[mg * CBIG + c] = acc[r] + bias;
    }
  }
}

extern "C" void kernel_launch(void* const* d_in, const int* in_sizes, int n_in,
                              void* d_out, int out_size, void* d_ws, size_t ws_size,
                              hipStream_t stream) {
  const float* x  = (const float*)d_in[0];
  const float* g1 = (const float*)d_in[1];
  const float* b1 = (const float*)d_in[2];
  const float* m1 = (const float*)d_in[3];
  const float* v1 = (const float*)d_in[4];
  const float* wq = (const float*)d_in[5];
  const float* bq = (const float*)d_in[6];
  const float* wk = (const float*)d_in[7];
  const float* bk = (const float*)d_in[8];
  const float* wv = (const float*)d_in[9];
  const float* bv = (const float*)d_in[10];
  const float* wp = (const float*)d_in[11];
  const float* bp = (const float*)d_in[12];
  const float* g2 = (const float*)d_in[13];
  const float* b2 = (const float*)d_in[14];
  const float* m2 = (const float*)d_in[15];
  const float* v2 = (const float*)d_in[16];

  char* ws = (char*)d_ws;
  _Float16* wqkv_frag = (_Float16*)(ws + OFF_WQKV);
  _Float16* wp_frag   = (_Float16*)(ws + OFF_WP);
  float*    bqkv      = (float*)(ws + OFF_BQKV);
  float*    bp2       = (float*)(ws + OFF_BP2);
  float*    sbuf      = (float*)(ws + OFF_S);
  _Float16* betaT     = (_Float16*)(ws + OFF_BETA);
  _Float16* qT        = (_Float16*)(ws + OFF_QT);
  _Float16* kT        = (_Float16*)(ws + OFF_KT);
  _Float16* vN        = (_Float16*)(ws + OFF_V);
  _Float16* attn      = (_Float16*)(ws + OFF_ATTN);

  k_prep<<<256, 256, 0, stream>>>(g1, b1, m1, v1, wq, bq, wk, bk, wv, bv, wp, bp,
                                  g2, b2, m2, v2, wqkv_frag, wp_frag, bqkv, bp2, sbuf);
  k_qkv<<<(BAT * NTOT) / 64, 128, 0, stream>>>(x, wqkv_frag, bqkv, qT, kT, vN);
  k_gram<<<BAT * 32, 128, 0, stream>>>(qT, kT, sbuf);
  k_softmax<<<(BAT * 64) / 4, 128, 0, stream>>>(sbuf, betaT);
  k_attn<<<BAT * 64, 128, 0, stream>>>(vN, betaT, attn);
  k_proj<<<(BAT * NTOT) / 64, 256, 0, stream>>>(attn, wp_frag, bp2, (float*)d_out);
}
